// GSageUnsup_45320494908016
// MI455X (gfx1250) — matrix operand order repaired and checked by reading, not compiled, not run on hardware
//
#include <hip/hip_runtime.h>

// ---------------------------------------------------------------------------
// GraphSAGE eval step on MI455X (gfx1250, wave32, WMMA).
// Key optimization: layer-1 (h1) is dead code w.r.t. the returned outputs,
// so we only compute h2 = LN(ReLU([mean(hidden1[neigh2]), hidden1[nb]] @ W2 + b2))
// and new_hidden2 = hidden2 with rows node_batch overwritten by h2.
// GEMM runs on v_wmma_f32_16x16x32_f16 (f16 inputs, f32 accumulate).
// ---------------------------------------------------------------------------

typedef __attribute__((ext_vector_type(16))) _Float16 v16h;
typedef __attribute__((ext_vector_type(8)))  float    v8f;

#define HDIM 256          // H2 (= hidden width, = b2/g2/be2 length)
#define KDIM 512          // 2*HDIM concat width
#define MT   16           // rows per block

// --- 1) W2 [512,256] f32  ->  Wt [256,512] f16 (transposed for contiguous B-frags)
__global__ __launch_bounds__(256) void wconv_kernel(const float* __restrict__ W,
                                                    _Float16* __restrict__ Wt) {
    int k = blockIdx.x;        // 0..511
    int n = threadIdx.x;       // 0..255
    Wt[(size_t)n * KDIM + k] = (_Float16)W[(size_t)k * HDIM + n];
}

// --- 2) hidden2 -> new_hidden2 region (bulk float4 copy)
__global__ __launch_bounds__(256) void copy4_kernel(const float4* __restrict__ src,
                                                    float4* __restrict__ dst, int n4) {
    int i = blockIdx.x * blockDim.x + threadIdx.x;
    if (i < n4) dst[i] = src[i];
}

// --- 3) build X [B,512] f16: [mean(hidden1[neigh2[b,:]]), hidden1[node_batch[b]]]
__global__ __launch_bounds__(256) void gather_kernel(const float* __restrict__ table,
                                                     const int* __restrict__ neigh,
                                                     const int* __restrict__ node_batch,
                                                     _Float16* __restrict__ Xh,
                                                     int S, float invS) {
    int b = blockIdx.x;
    int d = threadIdx.x;                       // 0..255
    const int* np = neigh + (size_t)b * S;
    float acc = 0.f;
    for (int s = 0; s < S; ++s) {
        int idx = np[s];
        acc += table[(size_t)idx * HDIM + d];  // coalesced 1KB row reads
    }
    acc *= invS;
    float selfv = table[(size_t)node_batch[b] * HDIM + d];
    _Float16* xr = Xh + (size_t)b * KDIM;
    xr[d]        = (_Float16)acc;
    xr[HDIM + d] = (_Float16)selfv;
}

// --- 4) fused GEMM + bias + ReLU + LayerNorm + scatter
__global__ __launch_bounds__(256) void gemm_ln_kernel(const _Float16* __restrict__ Xh,
                                                      const _Float16* __restrict__ Wt,
                                                      const float* __restrict__ bias,
                                                      const float* __restrict__ gamma,
                                                      const float* __restrict__ beta,
                                                      const int* __restrict__ node_batch,
                                                      float* __restrict__ out_h2,
                                                      float* __restrict__ out_tab) {
    __shared__ __align__(16) _Float16 lx[MT * KDIM];   // 16 KB: X tile (16 rows x 512)
    __shared__ __align__(16) float    lh[MT * HDIM];   // 16 KB: post-ReLU activations
    __shared__ float psum[MT][16];
    __shared__ float psq[MT][16];
    __shared__ float mus[MT], rss[MT];

    const int tid = threadIdx.x;
    const int m0  = blockIdx.x * MT;

    // stage X tile into LDS (coalesced, 1024 x 16B)
    {
        const uint4* src = (const uint4*)(Xh + (size_t)m0 * KDIM);
        uint4* dst = (uint4*)lx;
        for (int i = tid; i < (MT * KDIM) / 8; i += 256) dst[i] = src[i];
    }
    __syncthreads();

    const int lane   = tid & 31;        // wave32
    const int wave   = tid >> 5;        // 0..7
    const int n0     = wave * 32;       // each wave owns 32 output cols
    const int half16 = lane >> 4;       // lane-half selects K group
    const int l16    = lane & 15;

    v8f c0 = {0.f,0.f,0.f,0.f,0.f,0.f,0.f,0.f};
    v8f c1 = {0.f,0.f,0.f,0.f,0.f,0.f,0.f,0.f};

    // ISA layouts (CDNA5 ISA 7.12.2):
    //  A 16x32 f16: lane row M=l16; halves e0..7 -> K = 8*half16 + e,
    //               e8..15 -> K = 8*half16 + 16 + (e-8)  (two contiguous b128 runs)
    //  B 32x16 f16: lane col N=l16; same K mapping as A (two contiguous b128 runs)
    const int aoff = half16 * 8;
    const int boff = half16 * 8;
    const _Float16* arow = lx + (size_t)l16 * KDIM;
    const _Float16* b0p  = Wt + (size_t)(n0 + l16) * KDIM;
    const _Float16* b1p  = Wt + (size_t)(n0 + 16 + l16) * KDIM;

    for (int k0 = 0; k0 < KDIM; k0 += 32) {
        union { v16h v; uint4 q[2]; } fa, fb0, fb1;
        fa.q[0]  = *(const uint4*)(arow + k0 + aoff);        // ds_load_b128
        fa.q[1]  = *(const uint4*)(arow + k0 + aoff + 16);
        fb0.q[0] = *(const uint4*)(b0p + k0 + boff);         // global b128 (L2-hot)
        fb0.q[1] = *(const uint4*)(b0p + k0 + boff + 16);
        fb1.q[0] = *(const uint4*)(b1p + k0 + boff);
        fb1.q[1] = *(const uint4*)(b1p + k0 + boff + 16);
        c0 = __builtin_amdgcn_wmma_f32_16x16x32_f16(false, fa.v, false, fb0.v,
                                                    (short)0, c0, false, false);
        c1 = __builtin_amdgcn_wmma_f32_16x16x32_f16(false, fa.v, false, fb1.v,
                                                    (short)0, c1, false, false);
    }

    // bias + ReLU -> LDS (C/D layout: elem i -> M = i + 8*half16, N = l16)
    const float bb0 = bias[n0 + l16];
    const float bb1 = bias[n0 + 16 + l16];
    for (int i = 0; i < 8; ++i) {
        int m = i + half16 * 8;
        float v0 = c0[i] + bb0; v0 = v0 > 0.f ? v0 : 0.f;
        float v1 = c1[i] + bb1; v1 = v1 > 0.f ? v1 : 0.f;
        lh[m * HDIM + n0 + l16]      = v0;
        lh[m * HDIM + n0 + 16 + l16] = v1;
    }
    __syncthreads();

    // LayerNorm: 16 threads per row, 16 cols each
    {
        int r = tid >> 4, s = tid & 15;
        float su = 0.f, sq = 0.f;
        const float* row = lh + r * HDIM + s * 16;
        for (int c = 0; c < 16; ++c) { float x = row[c]; su += x; sq += x * x; }
        psum[r][s] = su; psq[r][s] = sq;
    }
    __syncthreads();
    if (tid < MT) {
        float su = 0.f, sq = 0.f;
        for (int s = 0; s < 16; ++s) { su += psum[tid][s]; sq += psq[tid][s]; }
        float mu  = su * (1.f / HDIM);
        float var = sq * (1.f / HDIM) - mu * mu;
        mus[tid] = mu;
        rss[tid] = rsqrtf(var + 1e-5f);
    }
    __syncthreads();

    // normalize + write h2 and scatter into new_hidden2
    const float g  = gamma[tid];
    const float be = beta[tid];
    for (int r = 0; r < MT; ++r) {
        float x = lh[r * HDIM + tid];
        float y = g * (x - mus[r]) * rss[r] + be;
        int row = m0 + r;
        out_h2[(size_t)row * HDIM + tid] = y;
        out_tab[(size_t)node_batch[row] * HDIM + tid] = y;
    }
}

extern "C" void kernel_launch(void* const* d_in, const int* in_sizes, int n_in,
                              void* d_out, int out_size, void* d_ws, size_t ws_size,
                              hipStream_t stream) {
    // setup_inputs() order:
    // 0 feats, 1 hidden1, 2 hidden2, 3 W1, 4 b1, 5 g1, 6 be1,
    // 7 W2, 8 b2, 9 g2, 10 be2, 11 node_batch, 12 neigh1, 13 neigh2
    const float* hidden1 = (const float*)d_in[1];
    const float* hidden2 = (const float*)d_in[2];
    const float* W2      = (const float*)d_in[7];
    const float* b2      = (const float*)d_in[8];
    const float* g2      = (const float*)d_in[9];
    const float* be2     = (const float*)d_in[10];
    const int* node_batch = (const int*)d_in[11];
    const int* neigh2     = (const int*)d_in[13];

    const int B  = in_sizes[11];          // 16384
    const int S  = in_sizes[13] / B;      // 10
    const int NH = in_sizes[2];           // N * 256

    char* ws = (char*)d_ws;
    _Float16* Wt = (_Float16*)ws;                                   // 256 KB
    _Float16* Xh = (_Float16*)(ws + (size_t)HDIM * KDIM * sizeof(_Float16));

    float* out_h2  = (float*)d_out;                 // [B,256]
    float* out_tab = out_h2 + (size_t)B * HDIM;     // [N,256] new_hidden2

    wconv_kernel<<<KDIM, HDIM, 0, stream>>>(W2, Wt);

    int n4 = NH / 4;
    copy4_kernel<<<(n4 + 255) / 256, 256, 0, stream>>>((const float4*)hidden2,
                                                       (float4*)out_tab, n4);

    gather_kernel<<<B, HDIM, 0, stream>>>(hidden1, neigh2, node_batch, Xh,
                                          S, 1.0f / (float)S);

    gemm_ln_kernel<<<B / MT, 256, 0, stream>>>(Xh, Wt, b2, g2, be2, node_batch,
                                               out_h2, out_tab);
}
